// GATEvolutionModel_83751862272246
// MI455X (gfx1250) — matrix operand order repaired and checked by reading, not compiled, not run on hardware
//
#include <hip/hip_runtime.h>
#include <hip/hip_bf16.h>

// ---------------------------------------------------------------------------
// GAT (2x GATConv + linear head) for MI455X / gfx1250.
// Dense projections: v_wmma_f32_16x16x32_bf16, one wave per 16-row strip
// computing ALL output columns (A fragment reuse; X read exactly once).
// W is pre-packed into WMMA B-fragment order; all B fragments for a k-step
// are preloaded as one clause, then the WMMA chain runs back-to-back.
// Edge softmax/aggregation: f32 atomics (bandwidth bound ~1.3 GB -> ~100us
// floor at 23.3 TB/s; GEMMs ~5 GFLOP, negligible on the WMMA path).
// ---------------------------------------------------------------------------

typedef __attribute__((ext_vector_type(16))) __bf16 v16bf;
typedef __attribute__((ext_vector_type(8)))  float  v8f;

#define FIN    128
#define H1     4
#define C1     32
#define HC1    128   // H1*C1
#define C2     64

static inline long cdivl(long a, long b) { return (a + b - 1) / b; }

// ---------------------------------------------------------------------------
__global__ void fill_u32(unsigned* __restrict__ p, long n, unsigned bits) {
    long i = (long)blockIdx.x * blockDim.x + threadIdx.x;
    if (i < n) p[i] = bits;
}

// ---------------------------------------------------------------------------
// Per-destination degree and edge_attr sums (self-loop fill_value='mean').
// ---------------------------------------------------------------------------
__global__ void deg_kernel(const int* __restrict__ dst, const float* __restrict__ ea,
                           float* __restrict__ deg, float* __restrict__ sat, int E) {
    int e = blockIdx.x * blockDim.x + threadIdx.x;
    if (e >= E) return;
    int d = dst[e];
    atomicAdd(&deg[d], 1.0f);
    atomicAdd(&sat[d], ea[e]);
}

__global__ void loopattr_kernel(const float* __restrict__ deg, const float* __restrict__ sat,
                                float* __restrict__ la, int n) {
    int i = blockIdx.x * blockDim.x + threadIdx.x;
    if (i < n) la[i] = sat[i] / fmaxf(deg[i], 1.0f);
}

// ---------------------------------------------------------------------------
// wedot[h] = sum_c We[h*C+c]*ae[h*C+c]  (lin_edge in_dim=1 =>
// a_edge[e,h] = ea[e] * wedot[h]).  wedot[0..3]=layer1, wedot[4]=layer2.
// ---------------------------------------------------------------------------
__global__ void prep_wedot(const float* __restrict__ We1, const float* __restrict__ ae1,
                           const float* __restrict__ We2, const float* __restrict__ ae2,
                           float* __restrict__ wedot) {
    int tid = blockIdx.x * blockDim.x + threadIdx.x;
    if (tid < H1) {
        float s = 0.f;
        for (int c = 0; c < C1; ++c) s += We1[tid * C1 + c] * ae1[tid * C1 + c];
        wedot[tid] = s;
    }
    if (tid == H1) {
        float s = 0.f;
        for (int c = 0; c < C2; ++c) s += We2[c] * ae2[c];
        wedot[H1] = s;
    }
}

// ---------------------------------------------------------------------------
// Pack W[128, Nout] (f32 row-major) into bf16 WMMA B-fragment order:
//   Wp[((tn*4 + ks)*32 + lane)*16 + i]
//   hi = lane>>4, col = tn*16 + (lane&15),
//   k  = ks*32 + hi*16 + i                (contiguous k order, same as the A loads)
// Each lane's 16 bf16 fragment is then a single contiguous 32-byte load.
// ---------------------------------------------------------------------------
__global__ void pack_W(const float* __restrict__ W, __bf16* __restrict__ Wp, int Nout) {
    int tid = blockIdx.x * blockDim.x + threadIdx.x;
    if (tid >= Nout * FIN) return;
    int i    = tid & 15;
    int lane = (tid >> 4) & 31;
    int ks   = (tid >> 9) & 3;
    int tn   = tid >> 11;
    int hi   = lane >> 4;
    int col  = tn * 16 + (lane & 15);
    int k    = ks * 32 + hi * 16 + i;
    Wp[tid] = (__bf16)W[k * Nout + col];
}

// ---------------------------------------------------------------------------
// Strip GEMM: Hout[N, NTILES*16] = X[N, 128] * W  (bf16 WMMA, f32 acc)
// One wave per 16-row strip; A fragment loaded once per k-step, reused for
// all NTILES column tiles. N multiple of 16 -> EXEC all-ones (WMMA req).
//
// A 16x32 bf16 lane layout used here (contiguous k, matches pack_W):
//   lane l<16 : row M=l,    K[0..15]
//   lane l>=16: row M=l-16, K[16..31]
// D 16x16 f32: col = lane&15; VGPR v = row v (lanes 0-15) / row 8+v (16-31)
// ---------------------------------------------------------------------------
template <int NTILES>
__global__ void gemm_wmma_strip(const float* __restrict__ X, const __bf16* __restrict__ Wp,
                                float* __restrict__ Hout) {
    const int Nout = NTILES * 16;
    const int tm   = blockIdx.x;
    const int lane = threadIdx.x;        // 0..31
    const int l15  = lane & 15;
    const int hi   = lane >> 4;

    const float* xr = X + (long)(tm * 16 + l15) * FIN;

    v8f acc[NTILES];
    #pragma unroll
    for (int t = 0; t < NTILES; ++t) acc[t] = (v8f){};

    #pragma unroll
    for (int ks = 0; ks < 4; ++ks) {
        const int k0 = ks * 32;

        // Preload ALL B fragments for this k-step (one clause of b128 loads).
        v16bf b[NTILES];
        #pragma unroll
        for (int t = 0; t < NTILES; ++t)
            b[t] = *(const v16bf*)(Wp + (((long)t * 4 + ks) * 32 + lane) * 16);

        // A fragment: two contiguous 8-float runs, converted to bf16.
        v16bf a;
        #pragma unroll
        for (int i = 0; i < 8; ++i) {
            a[i]     = (__bf16)xr[k0 + hi * 16 + i];       // K = k0 + hi*16 + i
            a[8 + i] = (__bf16)xr[k0 + hi * 16 + 8 + i];   // K = k0+hi*16+8+i
        }

        // Back-to-back WMMA chain, A reused across all column tiles.
        #pragma unroll
        for (int t = 0; t < NTILES; ++t)
            acc[t] = __builtin_amdgcn_wmma_f32_16x16x32_bf16(
                /*neg_a=*/false, a, /*neg_b=*/false, b[t],
                /*c_mod=*/(short)0, acc[t], /*reuse_a=*/false, /*reuse_b=*/false);
    }

    #pragma unroll
    for (int t = 0; t < NTILES; ++t) {
        #pragma unroll
        for (int v = 0; v < 8; ++v)
            Hout[(long)(tm * 16 + hi * 8 + v) * Nout + t * 16 + l15] = acc[t][v];
    }
}

// ---------------------------------------------------------------------------
// a_src[n,h] = <h[n,h,:], a_s[h,:]>,  a_dst likewise.
// ---------------------------------------------------------------------------
__global__ void att_dots(const float* __restrict__ hf, const float* __restrict__ a_s,
                         const float* __restrict__ a_d, float* __restrict__ asrc,
                         float* __restrict__ adst, int n, int H, int C) {
    int tid = blockIdx.x * blockDim.x + threadIdx.x;
    if (tid >= n * H) return;
    int node = tid / H, h = tid % H;
    const float* hp = hf + (long)node * H * C + h * C;
    float ss = 0.f, sd = 0.f;
    for (int c = 0; c < C; ++c) { ss += hp[c] * a_s[h * C + c]; sd += hp[c] * a_d[h * C + c]; }
    asrc[tid] = ss;
    adst[tid] = sd;
}

__device__ inline float lrelu02(float x) { return x > 0.f ? x : 0.2f * x; }

// float atomic max via signed-max / unsigned-min trick (init = -inf bits).
__device__ inline void atomicMaxF(float* addr, float v) {
    if (v >= 0.f) atomicMax((int*)addr, __float_as_int(v));
    else          atomicMin((unsigned int*)addr, __float_as_uint(v));
}

__device__ inline void edge_src_dst_attr(int e, int E, const int* src, const int* dst,
                                         const float* ea, const float* la,
                                         int& s, int& d, float& a) {
    if (e < E) { s = src[e]; d = dst[e]; a = ea[e]; }
    else       { s = d = e - E; a = la[s]; }          // self loop, mean attr
}

// ---------------------------------------------------------------------------
// Pass 1 of scatter-softmax: segment max of leaky_relu logits over dst.
// ---------------------------------------------------------------------------
__global__ void alpha_max_kernel(const float* __restrict__ asrc, const float* __restrict__ adst,
                                 const float* __restrict__ ea, const float* __restrict__ la,
                                 const float* __restrict__ wedot,
                                 const int* __restrict__ src, const int* __restrict__ dst,
                                 float* __restrict__ amax, int Etot, int E, int H) {
    long tid = (long)blockIdx.x * blockDim.x + threadIdx.x;
    if (tid >= (long)Etot * H) return;
    int e = (int)(tid / H), h = (int)(tid % H);
    int s, d; float a;
    edge_src_dst_attr(e, E, src, dst, ea, la, s, d, a);
    float alpha = lrelu02(asrc[(long)s * H + h] + adst[(long)d * H + h] + a * wedot[h]);
    atomicMaxF(&amax[(long)d * H + h], alpha);
}

// ---------------------------------------------------------------------------
// Pass 2: ex = exp(alpha - amax[dst]); store ex, accumulate denom[dst].
// ---------------------------------------------------------------------------
__global__ void alpha_exp_kernel(const float* __restrict__ asrc, const float* __restrict__ adst,
                                 const float* __restrict__ ea, const float* __restrict__ la,
                                 const float* __restrict__ wedot,
                                 const int* __restrict__ src, const int* __restrict__ dst,
                                 const float* __restrict__ amax, float* __restrict__ ex,
                                 float* __restrict__ denom, int Etot, int E, int H) {
    long tid = (long)blockIdx.x * blockDim.x + threadIdx.x;
    if (tid >= (long)Etot * H) return;
    int e = (int)(tid / H), h = (int)(tid % H);
    int s, d; float a;
    edge_src_dst_attr(e, E, src, dst, ea, la, s, d, a);
    float alpha = lrelu02(asrc[(long)s * H + h] + adst[(long)d * H + h] + a * wedot[h]);
    float v = __expf(alpha - amax[(long)d * H + h]);
    ex[tid] = v;
    atomicAdd(&denom[(long)d * H + h], v);
}

// ---------------------------------------------------------------------------
// out[dst, h, c] += h[src, h, c] * ex / (denom[dst,h] + 1e-16)
// One thread per (edge, channel); channels consecutive within a wave.
// ---------------------------------------------------------------------------
__global__ void aggregate_kernel(const float* __restrict__ hf, const float* __restrict__ ex,
                                 const float* __restrict__ denom,
                                 const int* __restrict__ src, const int* __restrict__ dst,
                                 float* __restrict__ out, int Etot, int E, int H, int C) {
    long tid = (long)blockIdx.x * blockDim.x + threadIdx.x;
    int HC = H * C;
    if (tid >= (long)Etot * HC) return;
    int e = (int)(tid / HC), c = (int)(tid % HC), h = c / C;
    int s, d;
    if (e < E) { s = src[e]; d = dst[e]; } else { s = d = e - E; }
    float att = ex[(long)e * H + h] / (denom[(long)d * H + h] + 1e-16f);
    atomicAdd(&out[(long)d * HC + c], hf[(long)s * HC + c] * att);
}

// out = elu(out + bias)   (in place)
__global__ void elu_bias_kernel(float* __restrict__ out, const float* __restrict__ b,
                                long n, int C) {
    long i = (long)blockIdx.x * blockDim.x + threadIdx.x;
    if (i >= n) return;
    float v = out[i] + b[i % C];
    out[i] = v > 0.f ? v : (__expf(v) - 1.f);
}

// y[n] = sum_c (out2[n,c] + b2[c]) * fcW[c] + fcb
__global__ void final_fc(const float* __restrict__ out2, const float* __restrict__ b2,
                         const float* __restrict__ fcW, const float* __restrict__ fcb,
                         float* __restrict__ y, int n, int C) {
    int i = blockIdx.x * blockDim.x + threadIdx.x;
    if (i >= n) return;
    float s = fcb[0];
    const float* op = out2 + (long)i * C;
    for (int c = 0; c < C; ++c) s += (op[c] + b2[c]) * fcW[c];
    y[i] = s;
}

// ---------------------------------------------------------------------------
extern "C" void kernel_launch(void* const* d_in, const int* in_sizes, int n_in,
                              void* d_out, int out_size, void* d_ws, size_t ws_size,
                              hipStream_t stream) {
    const float* x   = (const float*)d_in[0];
    const int*   ei  = (const int*)d_in[1];
    const float* ea  = (const float*)d_in[2];
    const float* W1  = (const float*)d_in[3];
    const float* as1 = (const float*)d_in[4];
    const float* ad1 = (const float*)d_in[5];
    const float* We1 = (const float*)d_in[6];
    const float* ae1 = (const float*)d_in[7];
    const float* b1  = (const float*)d_in[8];
    const float* W2  = (const float*)d_in[9];
    const float* as2 = (const float*)d_in[10];
    const float* ad2 = (const float*)d_in[11];
    const float* We2 = (const float*)d_in[12];
    const float* ae2 = (const float*)d_in[13];
    const float* b2  = (const float*)d_in[14];
    const float* fcW = (const float*)d_in[15];
    const float* fcb = (const float*)d_in[16];

    const int E = in_sizes[2];          // 1600000
    const int N = in_sizes[0] / FIN;    // 100000 (multiple of 16)
    const int Etot = E + N;
    const int* src = ei;
    const int* dst = ei + E;

    // ---- workspace layout (floats; all offsets 32B-aligned) ----
    float* ws = (float*)d_ws;
    size_t o = 0;
    float* h_buf   = ws + o; o += (size_t)N * FIN;   // h1; reused as h2 (N*64)
    float* out_buf = ws + o; o += (size_t)N * FIN;   // out1; reused as out2 (N*64)
    float* asrc    = ws + o; o += (size_t)N * H1;
    float* adst    = ws + o; o += (size_t)N * H1;
    float* amax    = ws + o; o += (size_t)N * H1;
    float* denom   = ws + o; o += (size_t)N * H1;
    float* exb     = ws + o; o += (size_t)Etot * H1;
    float* deg     = ws + o; o += (size_t)N;
    float* sat     = ws + o; o += (size_t)N;
    float* la      = ws + o; o += (size_t)N;
    float* wedot   = ws + o; o += 8;
    __bf16* W1p = (__bf16*)(ws + o); o += (FIN * HC1) / 2;
    __bf16* W2p = (__bf16*)(ws + o); o += (HC1 * C2) / 2;

    const int B = 256;
    const unsigned NEGINF = 0xFF800000u;

    // ---- self-loop mean edge_attr ----
    fill_u32<<<(int)cdivl(2L * N, B), B, 0, stream>>>((unsigned*)deg, 2L * N, 0u); // deg+sat
    deg_kernel<<<(int)cdivl(E, B), B, 0, stream>>>(dst, ea, deg, sat, E);
    loopattr_kernel<<<(int)cdivl(N, B), B, 0, stream>>>(deg, sat, la, N);

    // ---- weight prep: B-fragment packing + We.ae dots ----
    pack_W<<<(int)cdivl(FIN * HC1, B), B, 0, stream>>>(W1, W1p, HC1);
    pack_W<<<(int)cdivl(HC1 * C2, B), B, 0, stream>>>(W2, W2p, C2);
    prep_wedot<<<1, 32, 0, stream>>>(We1, ae1, We2, ae2, wedot);

    // ================= layer 1 (H=4, C=32) =================
    gemm_wmma_strip<HC1 / 16><<<N / 16, 32, 0, stream>>>(x, W1p, h_buf);
    att_dots<<<(int)cdivl((long)N * H1, B), B, 0, stream>>>(h_buf, as1, ad1, asrc, adst, N, H1, C1);

    fill_u32<<<(int)cdivl((long)N * H1, B), B, 0, stream>>>((unsigned*)amax, (long)N * H1, NEGINF);
    fill_u32<<<(int)cdivl((long)N * H1, B), B, 0, stream>>>((unsigned*)denom, (long)N * H1, 0u);
    fill_u32<<<(int)cdivl((long)N * FIN, B), B, 0, stream>>>((unsigned*)out_buf, (long)N * FIN, 0u);

    alpha_max_kernel<<<(int)cdivl((long)Etot * H1, B), B, 0, stream>>>(
        asrc, adst, ea, la, wedot, src, dst, amax, Etot, E, H1);
    alpha_exp_kernel<<<(int)cdivl((long)Etot * H1, B), B, 0, stream>>>(
        asrc, adst, ea, la, wedot, src, dst, amax, exb, denom, Etot, E, H1);
    aggregate_kernel<<<(int)cdivl((long)Etot * HC1, B), B, 0, stream>>>(
        h_buf, exb, denom, src, dst, out_buf, Etot, E, H1, C1);

    elu_bias_kernel<<<(int)cdivl((long)N * HC1, B), B, 0, stream>>>(out_buf, b1, (long)N * HC1, HC1);

    // ================= layer 2 (H=1, C=64) =================
    gemm_wmma_strip<C2 / 16><<<N / 16, 32, 0, stream>>>(out_buf, W2p, h_buf);
    att_dots<<<(int)cdivl(N, B), B, 0, stream>>>(h_buf, as2, ad2, asrc, adst, N, 1, C2);

    fill_u32<<<(int)cdivl(N, B), B, 0, stream>>>((unsigned*)amax, N, NEGINF);
    fill_u32<<<(int)cdivl(N, B), B, 0, stream>>>((unsigned*)denom, N, 0u);
    fill_u32<<<(int)cdivl((long)N * C2, B), B, 0, stream>>>((unsigned*)out_buf, (long)N * C2, 0u);

    float* wedot2 = wedot + H1;
    alpha_max_kernel<<<(int)cdivl(Etot, B), B, 0, stream>>>(
        asrc, adst, ea, la, wedot2, src, dst, amax, Etot, E, 1);
    alpha_exp_kernel<<<(int)cdivl(Etot, B), B, 0, stream>>>(
        asrc, adst, ea, la, wedot2, src, dst, amax, exb, denom, Etot, E, 1);
    aggregate_kernel<<<(int)cdivl((long)Etot * C2, B), B, 0, stream>>>(
        h_buf, exb, denom, src, dst, out_buf, Etot, E, 1, C2);

    // ================= head =================
    final_fc<<<(int)cdivl(N, B), B, 0, stream>>>(out_buf, b2, fcW, fcb, (float*)d_out, N, C2);
}
